// IterativeMessagePassing_8675833938120
// MI455X (gfx1250) — matrix operand order repaired and checked by reading, not compiled, not run on hardware
//
#include <hip/hip_runtime.h>
#include <math.h>

// Problem constants (from reference)
#define Bb  2
#define Cc  768
#define Vv  512
#define Dd  256
#define Hh  8
#define HDm 32
#define Dff 1024
#define Tt  8
#define ATT_SCALE 0.17677669529663687f   // 32^-0.5

typedef __attribute__((ext_vector_type(16))) __bf16 v16bf;
typedef __attribute__((ext_vector_type(8)))  __bf16 v8bf;
typedef __attribute__((ext_vector_type(8)))  float  v8f;
typedef __attribute__((ext_vector_type(4)))  int    v4i;

#if defined(__has_builtin)
#  if __has_builtin(__builtin_amdgcn_global_load_async_to_lds_b128) && \
      __has_builtin(__builtin_amdgcn_s_wait_asynccnt)
#    define USE_ASYNC_LDS 1
#  endif
#endif
#ifndef USE_ASYNC_LDS
#  define USE_ASYNC_LDS 0
#endif

#if USE_ASYNC_LDS
typedef __attribute__((address_space(1))) v4i as1_v4i;
typedef __attribute__((address_space(3))) v4i as3_v4i;
#endif

__device__ __forceinline__ v8f wmma_bf16(v16bf a, v16bf b, v8f c) {
  return __builtin_amdgcn_wmma_f32_16x16x32_bf16(false, a, false, b, (short)0, c,
                                                 false, false);
}
__device__ __forceinline__ v16bf cat8(v8bf a, v8bf b) {
  return __builtin_shufflevector(a, b, 0, 1, 2, 3, 4, 5, 6, 7, 8, 9, 10, 11, 12, 13, 14, 15);
}

// ---------------------------------------------------------------------------
// Elementwise helpers
// ---------------------------------------------------------------------------
__global__ void copy_k(const float* __restrict__ s, float* __restrict__ d, int n) {
  int i = blockIdx.x * blockDim.x + threadIdx.x;
  if (i < n) d[i] = s[i];
}
__global__ void zero_k(float* __restrict__ d, int n) {
  int i = blockIdx.x * blockDim.x + threadIdx.x;
  if (i < n) d[i] = 0.f;
}
__global__ void cvt_k(const float* __restrict__ s, __bf16* __restrict__ d, int n) {
  int i = blockIdx.x * blockDim.x + threadIdx.x;
  if (i < n) d[i] = (__bf16)s[i];
}

// LayerNorm over last dim (=256). One 256-thread block per row.
template <typename OutT>
__global__ void layernorm_k(const float* __restrict__ x, const float* __restrict__ g,
                            const float* __restrict__ b, OutT* __restrict__ y) {
  __shared__ float red[256];
  const int row = blockIdx.x, t = threadIdx.x;
  const float v = x[(size_t)row * Dd + t];
  red[t] = v; __syncthreads();
  for (int s = 128; s > 0; s >>= 1) { if (t < s) red[t] += red[t + s]; __syncthreads(); }
  const float mean = red[0] * (1.f / Dd);
  __syncthreads();
  const float d0 = v - mean;
  red[t] = d0 * d0; __syncthreads();
  for (int s = 128; s > 0; s >>= 1) { if (t < s) red[t] += red[t + s]; __syncthreads(); }
  const float inv = rsqrtf(red[0] * (1.f / Dd) + 1e-5f);
  y[(size_t)row * Dd + t] = (OutT)(d0 * inv * g[t] + b[t]);
}

// LSTM cell pointwise. gates f32, cell f32 in/out, h -> bf16 (GEMM input),
// state -> f32 (residual/LN chain).
__global__ void lstm_k(const float* __restrict__ gates, float* __restrict__ c,
                       __bf16* __restrict__ hb, float* __restrict__ state, int rows) {
  int idx = blockIdx.x * blockDim.x + threadIdx.x;
  if (idx >= rows * Dd) return;
  int row = idx >> 8, j = idx & 255;
  const float* gr = gates + (size_t)row * (4 * Dd);
  float ig = gr[j], fg = gr[Dd + j], gg = gr[2 * Dd + j], og = gr[3 * Dd + j];
  float si = 1.f / (1.f + __expf(-ig));
  float sf = 1.f / (1.f + __expf(-fg));
  float so = 1.f / (1.f + __expf(-og));
  float cn = sf * c[idx] + si * tanhf(gg);
  float hn = so * tanhf(cn);
  c[idx] = cn; hb[idx] = (__bf16)hn; state[idx] = hn;
}

// ---------------------------------------------------------------------------
// GEMM: Y[M,N] = act(X[M,K] @ W[N,K]^T + bias) (+= existing f32 Y if acc)
// X, W in bf16. Block = 128 threads (4 waves). Wave w owns 16x16 tile at
// (blockIdx.y*64 + 16w, blockIdx.x*16). The 16xK weight slab for this block's
// N-tile is staged once into LDS (async copy on CDNA5), shared by all 4 waves.
// outbf: 1 -> write bf16, 0 -> write f32 (acc only valid with f32).
// ---------------------------------------------------------------------------
__global__ void gemm_wmma(const __bf16* __restrict__ X, const __bf16* __restrict__ Wb,
                          const float* __restrict__ bias, void* __restrict__ Yv,
                          int M, int N, int K, int act, int acc, int outbf) {
  __shared__ __attribute__((aligned(16))) __bf16 Wt[16 * 1024];  // 16 x Kmax slab
  const int ntile = blockIdx.x * 16;

  // ---- cooperative stage of W slab [16 rows x K] into LDS ----
  const int chunks = (16 * K) / 8;  // 16-byte chunks; multiple of blockDim
  const int rowlen = K / 8;
  for (int cidx = threadIdx.x; cidx < chunks; cidx += blockDim.x) {
    const int wrow = cidx / rowlen;
    const int woff = (cidx % rowlen) * 8;
    const __bf16* g = Wb + (size_t)(ntile + wrow) * K + woff;
#if USE_ASYNC_LDS
    __builtin_amdgcn_global_load_async_to_lds_b128((as1_v4i*)g,
                                                   (as3_v4i*)&Wt[wrow * K + woff],
                                                   0, 0);
#else
    *(v8bf*)&Wt[wrow * K + woff] = *(const v8bf*)g;
#endif
  }
#if USE_ASYNC_LDS
  __builtin_amdgcn_s_wait_asynccnt(0);
#endif
  __syncthreads();

  const int lane = threadIdx.x & 31;
  const int wave = threadIdx.x >> 5;
  const bool hi  = lane >= 16;
  const int l15  = lane & 15;
  const int mtile = blockIdx.y * 64 + wave * 16;
  const int row  = mtile + l15;   // A-matrix row held by this lane
  const int ncol = ntile + l15;   // B-matrix column held by this lane

  v8f c = {};
  const __bf16* xr = X + (size_t)row * K + (hi ? 8 : 0);     // A: 2x 16B runs
  const __bf16* wr = &Wt[l15 * K + (hi ? 8 : 0)];            // B: 2x 16B runs (LDS)
  for (int kk = 0; kk < K; kk += 32) {
    __builtin_prefetch((const void*)(xr + kk + 128), 0, 1);
    v8bf a0 = *(const v8bf*)(xr + kk);
    v8bf a1 = *(const v8bf*)(xr + kk + 16);
    v8bf b0 = *(const v8bf*)(wr + kk);
    v8bf b1 = *(const v8bf*)(wr + kk + 16);
    c = wmma_bf16(cat8(a0, a1), cat8(b0, b1), c);
  }

  const float bn = bias ? bias[ncol] : 0.f;
#pragma unroll
  for (int r = 0; r < 8; ++r) {
    const int orow = mtile + (hi ? 8 + r : r);
    float v = c[r] + bn;
    if (act == 1) v = 0.5f * v * (1.f + erff(v * 0.70710678118654752f));  // exact GELU
    const size_t oi = (size_t)orow * N + ncol;
    if (outbf) {
      ((__bf16*)Yv)[oi] = (__bf16)v;
    } else {
      float* Y = (float*)Yv;
      if (acc) v += Y[oi];
      Y[oi] = v;
    }
  }
}

// ---------------------------------------------------------------------------
// Fused masked attention (flash-style, online softmax). bf16 in/out.
// One wave per (b, h, 16-row q tile). HD=32 => one WMMA K-step per score tile.
// Q/K/V/O: [B, S, D] bf16, head h at cols h*32..h*32+31.
// kvmask: [B, Sk] u8. sparse: vsm [B, Cc, Vv] f32 or nullptr.
//   strans=0: mask(q,k) = vsm[b, q, k];  strans=1: mask(q,k) = vsm[b, k, q]
// ---------------------------------------------------------------------------
__global__ void attn_k(const __bf16* __restrict__ Qm, const __bf16* __restrict__ Km,
                       const __bf16* __restrict__ Vm, __bf16* __restrict__ Om,
                       const unsigned char* __restrict__ kvmask,
                       const float* __restrict__ sparse, int strans,
                       int Sq, int Sk) {
  const int lane = threadIdx.x;
  const bool hi  = lane >= 16;
  const int l15  = lane & 15;
  const int qt   = blockIdx.x * 16;
  const int h    = blockIdx.y;
  const int b    = blockIdx.z;
  __shared__ __attribute__((aligned(16))) __bf16 Pt[16 * 32];

  const __bf16* qp = Qm + ((size_t)(b * Sq + qt + l15)) * Dd + h * HDm + (hi ? 8 : 0);
  const v16bf aq = cat8(*(const v8bf*)qp, *(const v8bf*)(qp + 16));

  float m[8], l[8];
  v8f acc0 = {}, acc1 = {};
#pragma unroll
  for (int r = 0; r < 8; ++r) { m[r] = -INFINITY; l[r] = 0.f; }

  for (int kk = 0; kk < Sk; kk += 32) {
    // ---- scores: two 16x16 WMMAs over this 32-key block ----
    const int kd = hi ? 8 : 0;
    const __bf16* kp0 = Km + ((size_t)(b * Sk + kk + l15)) * Dd + h * HDm + kd;
    const __bf16* kp1 = Km + ((size_t)(b * Sk + kk + 16 + l15)) * Dd + h * HDm + kd;
    const v16bf bk0 = cat8(*(const v8bf*)kp0, *(const v8bf*)(kp0 + 16));
    const v16bf bk1 = cat8(*(const v8bf*)kp1, *(const v8bf*)(kp1 + 16));
    v8f z = {};
    v8f s0 = wmma_bf16(aq, bk0, z);
    v8f s1 = wmma_bf16(aq, bk1, z);

    // ---- mask + online softmax (C layout: elem r <-> row, lane <-> col) ----
    float p0[8], p1[8];
    const int k0 = kk + l15, k1 = kk + 16 + l15;
    const bool km0 = kvmask[b * Sk + k0] != 0;
    const bool km1 = kvmask[b * Sk + k1] != 0;
#pragma unroll
    for (int r = 0; r < 8; ++r) {
      const int rowq = qt + (hi ? 8 + r : r);
      bool ok0 = km0, ok1 = km1;
      if (sparse) {
        const size_t o0 = strans ? ((size_t)(b * Cc + k0)) * Vv + rowq
                                 : ((size_t)(b * Cc + rowq)) * Vv + k0;
        const size_t o1 = strans ? ((size_t)(b * Cc + k1)) * Vv + rowq
                                 : ((size_t)(b * Cc + rowq)) * Vv + k1;
        ok0 = ok0 && (sparse[o0] != 0.f);
        ok1 = ok1 && (sparse[o1] != 0.f);
      }
      float v0 = ok0 ? s0[r] * ATT_SCALE : -INFINITY;
      float v1 = ok1 ? s1[r] * ATT_SCALE : -INFINITY;
      float t = fmaxf(v0, v1);
#pragma unroll
      for (int off = 1; off < 16; off <<= 1) t = fmaxf(t, __shfl_xor(t, off, 32));
      const float nm = fmaxf(m[r], t);
      const float sc = (m[r] > -INFINITY) ? __expf(m[r] - nm) : 0.f;
      const float e0 = (nm > -INFINITY) ? __expf(v0 - nm) : 0.f;
      const float e1 = (nm > -INFINITY) ? __expf(v1 - nm) : 0.f;
      float rs = e0 + e1;
#pragma unroll
      for (int off = 1; off < 16; off <<= 1) rs += __shfl_xor(rs, off, 32);
      l[r] = l[r] * sc + rs;
      acc0[r] *= sc; acc1[r] *= sc;
      m[r] = nm;
      p0[r] = e0; p1[r] = e1;
    }

    // ---- transpose P (C layout -> A layout) through LDS ----
    __syncthreads();
#pragma unroll
    for (int r = 0; r < 8; ++r) {
      const int prow = hi ? 8 + r : r;
      Pt[prow * 32 + l15]      = (__bf16)p0[r];
      Pt[prow * 32 + 16 + l15] = (__bf16)p1[r];
    }
    __syncthreads();
    const __bf16* pr = Pt + l15 * 32 + (hi ? 8 : 0);
    const v16bf ap = cat8(*(const v8bf*)pr, *(const v8bf*)(pr + 16));

    // ---- O += P @ V over this key block (two 16-col d-chunks) ----
    v16bf bv0, bv1;
#pragma unroll
    for (int e = 0; e < 16; ++e) {
      const int key = kk + (hi ? 8 : 0) + (e < 8 ? e : e + 8);
      const __bf16* vp = Vm + ((size_t)(b * Sk + key)) * Dd + h * HDm;
      bv0[e] = vp[l15];
      bv1[e] = vp[16 + l15];
    }
    acc0 = wmma_bf16(ap, bv0, acc0);
    acc1 = wmma_bf16(ap, bv1, acc1);
  }

  // ---- finalize: divide by row sum (empty rows -> 0, matches nan_to_num) ----
#pragma unroll
  for (int r = 0; r < 8; ++r) {
    const int rowq = qt + (hi ? 8 + r : r);
    const float inv = (l[r] > 0.f) ? 1.f / l[r] : 0.f;
    __bf16* op = Om + ((size_t)(b * Sq + rowq)) * Dd + h * HDm;
    op[l15]      = (__bf16)(acc0[r] * inv);
    op[16 + l15] = (__bf16)(acc1[r] * inv);
  }
}

// ---------------------------------------------------------------------------
// Host orchestration
// ---------------------------------------------------------------------------
extern "C" void kernel_launch(void* const* d_in, const int* in_sizes, int n_in,
                              void* d_out, int out_size, void* d_ws, size_t ws_size,
                              hipStream_t stream) {
  (void)in_sizes; (void)n_in; (void)out_size; (void)ws_size;
  auto F  = [&](int i) { return (const float*)d_in[i]; };
  auto M8 = [&](int i) { return (const unsigned char*)d_in[i]; };

  // Input index map (dict order, depth-first):
  // 0 clause_emb, 1 var_emb, 2 clause_mask, 3 var_mask, 4 vsm,
  // c2v(12) v2c(12) cff(6) vff(6) sa(16) clstm(4) vlstm(4), cfn_g/b, vfn_g/b
  const int C2V = 5, V2C = 17, CFF = 29, VFF = 35, SA = 41, CLSTM = 57, VLSTM = 61;

  const size_t NC = (size_t)Bb * Cc * Dd;   // 393216
  const size_t NV = (size_t)Bb * Vv * Dd;   // 262144

  // ---- workspace carve: f32 region then bf16 region ----
  float* fp = (float*)d_ws;
  auto takeF = [&](size_t n) { float* r = fp; fp += n; return r; };
  float* cs = takeF(NC);  float* vs = takeF(NV);
  float* cc = takeF(NC);  float* vc = takeF(NV);
  float* gates = takeF((size_t)Bb * Cc * Dff);

  __bf16* bp = (__bf16*)fp;
  auto takeB = [&](size_t n) { __bf16* r = bp; bp += n; return r; };
  __bf16* chb = takeB(NC);  __bf16* vhb = takeB(NV);
  __bf16* xq  = takeB(NC);  __bf16* xkv = takeB(NC);
  __bf16* qb  = takeB(NC);  __bf16* kb  = takeB(NC);  __bf16* vb = takeB(NC);
  __bf16* ab  = takeB(NC);  __bf16* upd = takeB(NC);
  __bf16* ffh = takeB((size_t)Bb * Cc * Dff);

  // ---- convert all weight matrices to bf16 once ----
  const __bf16* WB[70] = {};
  const int wIdx[22] = {C2V + 4, C2V + 6, C2V + 8, C2V + 10,
                        V2C + 4, V2C + 6, V2C + 8, V2C + 10,
                        SA + 4,  SA + 6,  SA + 8,  SA + 10,
                        CFF + 2, CFF + 4, VFF + 2, VFF + 4,
                        SA + 12, SA + 14,
                        CLSTM + 0, CLSTM + 2, VLSTM + 0, VLSTM + 2};
  for (int i = 0; i < 22; ++i) {
    const int n = (i < 12) ? Dd * Dd : Dff * Dd;
    __bf16* dst = takeB((size_t)n);
    cvt_k<<<n / 256, 256, 0, stream>>>(F(wIdx[i]), dst, n);
    WB[wIdx[i]] = dst;
  }

  auto LNb = [&](const float* x, const float* g, const float* be, __bf16* y, int rows) {
    layernorm_k<__bf16><<<rows, 256, 0, stream>>>(x, g, be, y);
  };
  auto LNf = [&](const float* x, const float* g, const float* be, float* y, int rows) {
    layernorm_k<float><<<rows, 256, 0, stream>>>(x, g, be, y);
  };
  auto GEMM = [&](const __bf16* X, const __bf16* W, const float* bias, void* Y,
                  int M, int N, int K, int act, int acc, int outbf) {
    gemm_wmma<<<dim3(N / 16, M / 64), 128, 0, stream>>>(X, W, bias, Y, M, N, K,
                                                        act, acc, outbf);
  };
  auto ATTN = [&](const __bf16* Q, const __bf16* K, const __bf16* Vp, __bf16* O,
                  const unsigned char* kvm, const float* sp, int strans, int Sq, int Sk) {
    attn_k<<<dim3(Sq / 16, Hh, Bb), 32, 0, stream>>>(Q, K, Vp, O, kvm, sp, strans, Sq, Sk);
  };
  auto LSTM = [&](const float* g, float* cbuf, __bf16* hbuf, float* st, int rows) {
    lstm_k<<<(rows * Dd) / 256, 256, 0, stream>>>(g, cbuf, hbuf, st, rows);
  };

  // ---- init states ----
  copy_k<<<(int)(NC / 256), 256, 0, stream>>>(F(0), cs, (int)NC);
  cvt_k <<<(int)(NC / 256), 256, 0, stream>>>(F(0), chb, (int)NC);
  zero_k<<<(int)(NC / 256), 256, 0, stream>>>(cc, (int)NC);
  copy_k<<<(int)(NV / 256), 256, 0, stream>>>(F(1), vs, (int)NV);
  cvt_k <<<(int)(NV / 256), 256, 0, stream>>>(F(1), vhb, (int)NV);
  zero_k<<<(int)(NV / 256), 256, 0, stream>>>(vc, (int)NV);

  for (int t = 0; t < Tt; ++t) {
    // ---- var update: cross-attn (q=vars, kv=clauses), params c2v ----
    LNb(vs, F(C2V + 0), F(C2V + 1), xq, Bb * Vv);
    LNb(cs, F(C2V + 2), F(C2V + 3), xkv, Bb * Cc);
    GEMM(xq,  WB[C2V + 4], F(C2V + 5), qb, Bb * Vv, Dd, Dd, 0, 0, 1);
    GEMM(xkv, WB[C2V + 6], F(C2V + 7), kb, Bb * Cc, Dd, Dd, 0, 0, 1);
    GEMM(xkv, WB[C2V + 8], F(C2V + 9), vb, Bb * Cc, Dd, Dd, 0, 0, 1);
    ATTN(qb, kb, vb, ab, M8(2), F(4), /*strans=*/1, Vv, Cc);
    GEMM(ab, WB[C2V + 10], F(C2V + 11), upd, Bb * Vv, Dd, Dd, 0, 0, 1);
    // var LSTM
    GEMM(upd, WB[VLSTM + 0], F(VLSTM + 1), gates, Bb * Vv, Dff, Dd, 0, 0, 0);
    GEMM(vhb, WB[VLSTM + 2], F(VLSTM + 3), gates, Bb * Vv, Dff, Dd, 0, 1, 0);
    LSTM(gates, vc, vhb, vs, Bb * Vv);
    // var FFN (residual)
    LNb(vs, F(VFF + 0), F(VFF + 1), xq, Bb * Vv);
    GEMM(xq,  WB[VFF + 2], F(VFF + 3), ffh, Bb * Vv, Dff, Dd, /*gelu*/1, 0, 1);
    GEMM(ffh, WB[VFF + 4], F(VFF + 5), vs,  Bb * Vv, Dd, Dff, 0, /*acc*/1, 0);

    // ---- clause update: cross-attn (q=clauses, kv=vars), params v2c ----
    LNb(cs, F(V2C + 0), F(V2C + 1), xq, Bb * Cc);
    LNb(vs, F(V2C + 2), F(V2C + 3), xkv, Bb * Vv);
    GEMM(xq,  WB[V2C + 4], F(V2C + 5), qb, Bb * Cc, Dd, Dd, 0, 0, 1);
    GEMM(xkv, WB[V2C + 6], F(V2C + 7), kb, Bb * Vv, Dd, Dd, 0, 0, 1);
    GEMM(xkv, WB[V2C + 8], F(V2C + 9), vb, Bb * Vv, Dd, Dd, 0, 0, 1);
    ATTN(qb, kb, vb, ab, M8(3), F(4), /*strans=*/0, Cc, Vv);
    GEMM(ab, WB[V2C + 10], F(V2C + 11), upd, Bb * Cc, Dd, Dd, 0, 0, 1);
    // clause LSTM
    GEMM(upd, WB[CLSTM + 0], F(CLSTM + 1), gates, Bb * Cc, Dff, Dd, 0, 0, 0);
    GEMM(chb, WB[CLSTM + 2], F(CLSTM + 3), gates, Bb * Cc, Dff, Dd, 0, 1, 0);
    LSTM(gates, cc, chb, cs, Bb * Cc);
    // clause FFN (residual)
    LNb(cs, F(CFF + 0), F(CFF + 1), xq, Bb * Cc);
    GEMM(xq,  WB[CFF + 2], F(CFF + 3), ffh, Bb * Cc, Dff, Dd, 1, 0, 1);
    GEMM(ffh, WB[CFF + 4], F(CFF + 5), cs,  Bb * Cc, Dd, Dff, 0, 1, 0);

    // ---- clause self-attention block ----
    LNb(cs, F(SA + 0), F(SA + 1), xq, Bb * Cc);
    GEMM(xq, WB[SA + 4], F(SA + 5), qb, Bb * Cc, Dd, Dd, 0, 0, 1);
    GEMM(xq, WB[SA + 6], F(SA + 7), kb, Bb * Cc, Dd, Dd, 0, 0, 1);
    GEMM(xq, WB[SA + 8], F(SA + 9), vb, Bb * Cc, Dd, Dd, 0, 0, 1);
    ATTN(qb, kb, vb, ab, M8(2), nullptr, 0, Cc, Cc);
    GEMM(ab, WB[SA + 10], F(SA + 11), cs, Bb * Cc, Dd, Dd, 0, /*acc*/1, 0);
    LNb(cs, F(SA + 2), F(SA + 3), xq, Bb * Cc);
    GEMM(xq,  WB[SA + 12], F(SA + 13), ffh, Bb * Cc, Dff, Dd, 1, 0, 1);
    GEMM(ffh, WB[SA + 14], F(SA + 15), cs,  Bb * Cc, Dd, Dff, 0, 1, 0);
  }

  // final LayerNorms straight into d_out: (clause, var) concatenated, f32
  float* out = (float*)d_out;
  LNf(cs, F(65), F(66), out, Bb * Cc);
  LNf(vs, F(67), F(68), out + NC, Bb * Vv);
}
